// FFA_90486370992428
// MI455X (gfx1250) — matrix operand order repaired and checked by reading, not compiled, not run on hardware
//
#include <hip/hip_runtime.h>
#include <hip/hip_bf16.h>

typedef _Float16 f16;
typedef __attribute__((ext_vector_type(16))) _Float16 v16h;
typedef __attribute__((ext_vector_type(8)))  _Float16 v8h;
typedef __attribute__((ext_vector_type(8)))  float    v8f;

#define HW 4096
#define CDIM 256

// ---- workspace layout (f16 elements) ----
constexpr size_t OFF_XHT = 0;                          // x^T  [4096][256]
constexpr size_t OFF_WQH = OFF_XHT + (size_t)HW*CDIM;  // wqkv [768][256]
constexpr size_t OFF_WPH = OFF_WQH + (size_t)768*CDIM; // wproj[256][256]
constexpr size_t OFF_QN  = OFF_WPH + (size_t)CDIM*CDIM;// q    [8][4096][32] (l2-normed)
constexpr size_t OFF_KN  = OFF_QN  + (size_t)8*HW*32;  // k    [8][4096][32] (l2-normed)
constexpr size_t OFF_VT  = OFF_KN  + (size_t)8*HW*32;  // v^T  [8][32][4096]
constexpr size_t OFF_OGT = OFF_VT  + (size_t)8*32*HW;  // attn-out in proj's [N=s2][K=c]

__device__ __forceinline__ float red_sum16(float v) {
#pragma unroll
  for (int m = 1; m < 16; m <<= 1) v += __shfl_xor(v, m, 32);
  return v;
}

// ---------------- stage 0: fp32 -> f16 conversions ----------------
__global__ __launch_bounds__(256) void cvt_f32_to_f16(const float* __restrict__ src,
                                                      f16* __restrict__ dst, int n) {
  int i = blockIdx.x * 256 + threadIdx.x;
  if (i < n) dst[i] = (f16)src[i];
}

// x (c=256, s=4096) -> xht[s][c]
__global__ __launch_bounds__(256) void transpose_x(const float* __restrict__ x,
                                                   f16* __restrict__ xht) {
  int i = blockIdx.x * 256 + threadIdx.x;   // i = s*256 + c
  int c = i & 255;
  int s = i >> 8;
  xht[i] = (f16)x[c * HW + s];
}

// ---------------- stage 1: QKV GEMM + fused l2norm ----------------
// grid: (64 s-blocks of 64, 24 row-blocks of 32).  8 waves, one 16x16 tile each.
__global__ __launch_bounds__(256) void qkv_gemm_norm(const f16* __restrict__ wqh,
                                                     const f16* __restrict__ xht,
                                                     f16* __restrict__ qn,
                                                     f16* __restrict__ kn,
                                                     f16* __restrict__ vt) {
  __shared__ float lds_ss[2][4][16];
  const int sb = blockIdx.x, rb = blockIdx.y;
  const int wv = threadIdx.x >> 5, lane = threadIdx.x & 31;
  const int l15 = lane & 15, hi = lane >> 4;
  const int tr = wv >> 2, tc = wv & 3;
  const int rowA = rb * 32 + tr * 16 + l15;      // A row (output row o)
  const int colB = sb * 64 + tc * 16 + l15;      // B col (spatial s)

  v8f acc = {};
#pragma unroll
  for (int k0 = 0; k0 < CDIM; k0 += 32) {
    v16h a, b;
    const f16* ap = wqh + rowA * CDIM + k0;
    ((uint4*)&a)[0] = *(const uint4*)(ap + 16 * hi);
    ((uint4*)&a)[1] = *(const uint4*)(ap + 16 * hi + 8);
    const f16* bp = xht + colB * CDIM + k0 + 16 * hi;
    ((uint4*)&b)[0] = *(const uint4*)(bp);
    ((uint4*)&b)[1] = *(const uint4*)(bp + 8);
    acc = __builtin_amdgcn_wmma_f32_16x16x32_f16(false, a, false, b, (short)0, acc,
                                                 false, false);
  }

  const int kind = rb >> 3;   // 0=q 1=k 2=v
  const int h = rb & 7;
  if (kind == 2) {            // v: store transposed [h][d][t]
#pragma unroll
    for (int r = 0; r < 8; ++r) {
      int d = tr * 16 + 8 * hi + r;
      vt[(h * 32 + d) * HW + colB] = (f16)acc[r];
    }
  } else {                    // q/k: l2norm over d (32 rows) per column s
    float ss = 0.f;
#pragma unroll
    for (int r = 0; r < 8; ++r) ss += acc[r] * acc[r];
    ss += __shfl_xor(ss, 16, 32);                 // combine both 8-row halves
    if (lane < 16) lds_ss[tr][tc][l15] = ss;      // exchange across the 2 row-tiles
    __syncthreads();
    float tot = lds_ss[0][tc][l15] + lds_ss[1][tc][l15];
    float scale = 1.0f / fmaxf(sqrtf(tot), 1e-12f);
    f16* dst = (kind == 0) ? qn : kn;
    v8h pv;
#pragma unroll
    for (int r = 0; r < 8; ++r) pv[r] = (f16)(acc[r] * scale);
    int dbase = tr * 16 + 8 * hi;                 // 8 consecutive d -> one b128 store
    *(uint4*)(dst + ((size_t)h * HW + colB) * 32 + dbase) = *(uint4*)&pv;
  }
}

// ---------------- stage 2: flash attention (static-max softmax) ----------------
// q,k are l2-normalized => |logit| <= |temp|.  Use M = |temp| as a static softmax
// max: exp(s-M) in [e^{-2|temp|}, 1] (temp=1 -> >= 0.135), so no running max, no
// rescale, and the row-sum reduces across lanes once at the end.
// grid: (32, 8 heads).  8 waves/WG; each wave owns 16 query rows, full 4096 keys.
__global__ __launch_bounds__(256) void attn_flash(const f16* __restrict__ qn,
                                                  const f16* __restrict__ kn,
                                                  const f16* __restrict__ vt,
                                                  const float* __restrict__ temperature,
                                                  f16* __restrict__ ogt) {
  __shared__ __align__(16) f16 plds[8][16][32];   // per-wave P tile (D->A relayout)
  const int h = blockIdx.y;
  const int wv = threadIdx.x >> 5, lane = threadIdx.x & 31;
  const int l15 = lane & 15, hi = lane >> 4;
  const int s0 = (blockIdx.x * 8 + wv) * 16;
  const float temp = temperature[h];
  const float M = fabsf(temp);

  // Q A-tile: rows s0..s0+15, K = d = 0..31
  v16h qa;
  {
    const f16* p = qn + ((size_t)h * HW + s0 + l15) * 32;
    ((uint4*)&qa)[0] = *(const uint4*)(p + 16 * hi);
    ((uint4*)&qa)[1] = *(const uint4*)(p + 16 * hi + 8);
  }

  float lpart[8];
  v8f o0 = {}, o1 = {};
#pragma unroll
  for (int r = 0; r < 8; ++r) lpart[r] = 0.f;

  const f16* kb = kn + (size_t)h * HW * 32;
  const f16* vb = vt + (size_t)h * 32 * HW;

#pragma unroll 2
  for (int t0 = 0; t0 < HW; t0 += 32) {
    // S = Q K^T : two 16x16 tiles (cols t0.., t0+16..)
    v16h k0t, k1t;
    const f16* kp0 = kb + (size_t)(t0 + l15) * 32 + 16 * hi;
    const f16* kp1 = kb + (size_t)(t0 + 16 + l15) * 32 + 16 * hi;
    ((uint4*)&k0t)[0] = *(const uint4*)(kp0);
    ((uint4*)&k0t)[1] = *(const uint4*)(kp0 + 8);
    ((uint4*)&k1t)[0] = *(const uint4*)(kp1);
    ((uint4*)&k1t)[1] = *(const uint4*)(kp1 + 8);
    v8f z = {};
    v8f sA = __builtin_amdgcn_wmma_f32_16x16x32_f16(false, qa, false, k0t, (short)0, z,
                                                    false, false);
    v8f sB = __builtin_amdgcn_wmma_f32_16x16x32_f16(false, qa, false, k1t, (short)0, z,
                                                    false, false);
    // P = exp(s*temp - M); accumulate per-lane row-sum partials (no shuffles here)
#pragma unroll
    for (int r = 0; r < 8; ++r) {
      float p0 = __expf(sA[r] * temp - M);
      float p1 = __expf(sB[r] * temp - M);
      lpart[r] += p0 + p1;
      plds[wv][8 * hi + r][l15] = (f16)p0;        // D-layout scatter into LDS
      plds[wv][8 * hi + r][16 + l15] = (f16)p1;
    }
#if __has_builtin(__builtin_amdgcn_wave_barrier)
    __builtin_amdgcn_wave_barrier();              // wave-private region: ordering only
#else
    __syncthreads();
#endif
    // reload P as a 16x32 A-tile
    v16h pa;
    const f16* pp = &plds[wv][l15][0];
    ((uint4*)&pa)[0] = *(const uint4*)(pp + 16 * hi);
    ((uint4*)&pa)[1] = *(const uint4*)(pp + 16 * hi + 8);
    // V B-tiles: K = t (32 rows), N = d (two 16-col tiles)
    v16h v0t, v1t;
    const f16* vp0 = vb + (size_t)l15 * HW + t0 + 16 * hi;
    const f16* vp1 = vb + (size_t)(16 + l15) * HW + t0 + 16 * hi;
    ((uint4*)&v0t)[0] = *(const uint4*)(vp0);
    ((uint4*)&v0t)[1] = *(const uint4*)(vp0 + 8);
    ((uint4*)&v1t)[0] = *(const uint4*)(vp1);
    ((uint4*)&v1t)[1] = *(const uint4*)(vp1 + 8);
    o0 = __builtin_amdgcn_wmma_f32_16x16x32_f16(false, pa, false, v0t, (short)0, o0,
                                                false, false);
    o1 = __builtin_amdgcn_wmma_f32_16x16x32_f16(false, pa, false, v1t, (short)0, o1,
                                                false, false);
  }

  // Epilogue. Reference does (b,s,heads,d).reshape(b,c,h,w): a flat reinterpret.
  // flat = s*256 + h*32 + d ; proj sees B[c][s2] with c = flat>>12, s2 = flat&4095.
  // Store OGT[s2][c] so the proj GEMM gets contiguous-K B rows.
#pragma unroll
  for (int r = 0; r < 8; ++r) {
    int s = s0 + 8 * hi + r;
    float inv = 1.0f / red_sum16(lpart[r]);       // row-sum reduction, once
    int flat0 = s * 256 + h * 32 + l15;
    int flat1 = flat0 + 16;
    ogt[(size_t)(flat0 & (HW - 1)) * CDIM + (flat0 >> 12)] = (f16)(o0[r] * inv);
    ogt[(size_t)(flat1 & (HW - 1)) * CDIM + (flat1 >> 12)] = (f16)(o1[r] * inv);
  }
}

// ---------------- stage 3: proj GEMM ----------------
// out[o][s2] = sum_c wproj[o][c] * B[c][s2], B^T rows = ogt.  grid: (64, 8).
__global__ __launch_bounds__(256) void proj_gemm(const f16* __restrict__ wph,
                                                 const f16* __restrict__ ogt,
                                                 float* __restrict__ out) {
  const int sb = blockIdx.x, rb = blockIdx.y;
  const int wv = threadIdx.x >> 5, lane = threadIdx.x & 31;
  const int l15 = lane & 15, hi = lane >> 4;
  const int tr = wv >> 2, tc = wv & 3;
  const int rowA = rb * 32 + tr * 16 + l15;
  const int colB = sb * 64 + tc * 16 + l15;

  v8f acc = {};
#pragma unroll
  for (int k0 = 0; k0 < CDIM; k0 += 32) {
    v16h a, b;
    const f16* ap = wph + rowA * CDIM + k0;
    ((uint4*)&a)[0] = *(const uint4*)(ap + 16 * hi);
    ((uint4*)&a)[1] = *(const uint4*)(ap + 16 * hi + 8);
    const f16* bp = ogt + (size_t)colB * CDIM + k0 + 16 * hi;
    ((uint4*)&b)[0] = *(const uint4*)(bp);
    ((uint4*)&b)[1] = *(const uint4*)(bp + 8);
    acc = __builtin_amdgcn_wmma_f32_16x16x32_f16(false, a, false, b, (short)0, acc,
                                                 false, false);
  }
  const int orow = rb * 32 + tr * 16 + 8 * hi;
#pragma unroll
  for (int r = 0; r < 8; ++r)
    out[(size_t)(orow + r) * HW + colB] = acc[r];
}

extern "C" void kernel_launch(void* const* d_in, const int* in_sizes, int n_in,
                              void* d_out, int out_size, void* d_ws, size_t ws_size,
                              hipStream_t stream) {
  const float* x     = (const float*)d_in[0];   // (1,256,64,64)
  const float* wqkv  = (const float*)d_in[1];   // (768,256)
  const float* wproj = (const float*)d_in[2];   // (256,256)
  const float* temp  = (const float*)d_in[3];   // (8,1,1)
  float* out = (float*)d_out;                   // (1,256,64,64)

  f16* ws  = (f16*)d_ws;
  f16* XHT = ws + OFF_XHT;
  f16* WQH = ws + OFF_WQH;
  f16* WPH = ws + OFF_WPH;
  f16* QN  = ws + OFF_QN;
  f16* KN  = ws + OFF_KN;
  f16* VT  = ws + OFF_VT;
  f16* OGT = ws + OFF_OGT;

  cvt_f32_to_f16<<<(768 * CDIM + 255) / 256, 256, 0, stream>>>(wqkv, WQH, 768 * CDIM);
  cvt_f32_to_f16<<<(CDIM * CDIM + 255) / 256, 256, 0, stream>>>(wproj, WPH, CDIM * CDIM);
  transpose_x<<<(HW * CDIM) / 256, 256, 0, stream>>>(x, XHT);

  qkv_gemm_norm<<<dim3(64, 24), 256, 0, stream>>>(WQH, XHT, QN, KN, VT);
  attn_flash<<<dim3(32, 8), 256, 0, stream>>>(QN, KN, VT, temp, OGT);
  proj_gemm<<<dim3(64, 8), 256, 0, stream>>>(WPH, OGT, out);
}
